// net_1425929142877
// MI455X (gfx1250) — matrix operand order repaired and checked by reading, not compiled, not run on hardware
//
#include <hip/hip_runtime.h>

typedef __attribute__((ext_vector_type(16))) _Float16 v16h;
typedef __attribute__((ext_vector_type(8)))  _Float16 v8h;
typedef __attribute__((ext_vector_type(8)))  float    v8f;
typedef __attribute__((__vector_size__(4 * sizeof(int)))) int vs4i;

#define KPAD 2048
#define NPAD 2048
#define MPAD 40064   // 40000 rows padded to a multiple of 128
#define BN_EPS 1e-5f
#define BSTRIDE 72   // 64 + 8 f16 pad -> 36 dwords, conflict-free ds_load_b128

#if defined(__AMDGCN__) && __has_builtin(__builtin_amdgcn_global_load_async_to_lds_b128)
#define HAVE_ASYNC_COPY 1
#else
#define HAVE_ASYNC_COPY 0
#endif

// 16-byte global -> LDS copy: async DMA (ASYNCcnt) if available, else via VGPRs
__device__ __forceinline__ void copy16_g2l(const _Float16* g, _Float16* l)
{
#if HAVE_ASYNC_COPY
    __builtin_amdgcn_global_load_async_to_lds_b128(
        (__attribute__((address_space(1))) vs4i*)(void*)g,
        (__attribute__((address_space(3))) vs4i*)(void*)l, 0, 0);
#else
    *(v8h*)l = *(const v8h*)g;
#endif
}

__device__ __forceinline__ void wait_async_all()
{
#if HAVE_ASYNC_COPY
# if __has_builtin(__builtin_amdgcn_s_wait_asynccnt)
    __builtin_amdgcn_s_wait_asynccnt(0);
# else
    asm volatile("s_wait_asynccnt 0x0" ::: "memory");
# endif
#endif
}

// ---------------------------------------------------------------------------
// Conv stem: 3x3, stride 2, pad 1 (direct), bias fused
// ---------------------------------------------------------------------------
__global__ void conv3x3_s2(const float* __restrict__ in, const float* __restrict__ w,
                           const float* __restrict__ cb, float* __restrict__ out,
                           int B, int Cin, int Hin, int Cout, int Hout)
{
    int idx = blockIdx.x * blockDim.x + threadIdx.x;
    int total = B * Cout * Hout * Hout;
    if (idx >= total) return;
    int x  = idx % Hout;
    int y  = (idx / Hout) % Hout;
    int co = (idx / (Hout * Hout)) % Cout;
    int b  = idx / (Hout * Hout * Cout);
    float acc = cb[co];
    for (int ci = 0; ci < Cin; ++ci) {
        const float* ip = in + ((b * Cin + ci) * Hin) * Hin;
        const float* wp = w + ((co * Cin + ci) * 3) * 3;
        #pragma unroll
        for (int dy = 0; dy < 3; ++dy) {
            int iy = 2 * y - 1 + dy;
            if (iy < 0 || iy >= Hin) continue;
            #pragma unroll
            for (int dx = 0; dx < 3; ++dx) {
                int ix = 2 * x - 1 + dx;
                if (ix < 0 || ix >= Hin) continue;
                acc += ip[iy * Hin + ix] * wp[dy * 3 + dx];
            }
        }
    }
    out[idx] = acc;
}

// Per-channel training-mode batchnorm stats (biased variance)
__global__ void bn_stats(const float* __restrict__ x, float* __restrict__ mean,
                         float* __restrict__ istd, int B, int C, int HW)
{
    __shared__ float s0[256], s1[256];
    int c = blockIdx.x;
    int n = B * HW;
    float sum = 0.f, sq = 0.f;
    for (int i = threadIdx.x; i < n; i += 256) {
        int b = i / HW, p = i % HW;
        float v = x[(b * C + c) * HW + p];
        sum += v; sq += v * v;
    }
    s0[threadIdx.x] = sum; s1[threadIdx.x] = sq;
    __syncthreads();
    for (int off = 128; off > 0; off >>= 1) {
        if (threadIdx.x < off) {
            s0[threadIdx.x] += s0[threadIdx.x + off];
            s1[threadIdx.x] += s1[threadIdx.x + off];
        }
        __syncthreads();
    }
    if (threadIdx.x == 0) {
        float mu  = s0[0] / (float)n;
        float var = s1[0] / (float)n - mu * mu;
        mean[c] = mu;
        istd[c] = rsqrtf(var + BN_EPS);
    }
}

__global__ void bn_relu(float* __restrict__ x, const float* __restrict__ mean,
                        const float* __restrict__ istd, const float* __restrict__ g,
                        const float* __restrict__ be, int B, int C, int HW)
{
    int idx = blockIdx.x * blockDim.x + threadIdx.x;
    int total = B * C * HW;
    if (idx >= total) return;
    int c = (idx / HW) % C;
    float v = (x[idx] - mean[c]) * istd[c] * g[c] + be[c];
    x[idx] = v > 0.f ? v : 0.f;
}

// ---------------------------------------------------------------------------
// feat[b*25 + obj][c] : 256 channels + 2 coord channels (linspace(-2.5,2.5,5))
// ---------------------------------------------------------------------------
__global__ void build_feat(const float* __restrict__ x4, float* __restrict__ feat)
{
    int idx = blockIdx.x * blockDim.x + threadIdx.x;
    int total = 64 * 25 * 258;
    if (idx >= total) return;
    int c   = idx % 258;
    int obj = (idx / 258) % 25;
    int b   = idx / (258 * 25);
    float v;
    if (c < 256) {
        v = x4[(b * 256 + c) * 25 + obj];
    } else {
        int xx = obj % 5, yy = obj / 5;
        int p = (c == 256) ? xx : yy;
        v = -2.5f + 1.25f * (float)p;
    }
    feat[idx] = v;
}

// ---------------------------------------------------------------------------
// Small fp32 GEMM (used for the factorized g-layer-1 pieces).
// out[r][n] (NPAD stride, zero-padded cols) = sum_k A[r*KA+k] * W[k*ldW+n]
// ---------------------------------------------------------------------------
__global__ void gemm_slice(const float* __restrict__ A, int R, int KA, int Krows,
                           const float* __restrict__ W, int ldW, int N,
                           float* __restrict__ out)
{
    int idx = blockIdx.x * blockDim.x + threadIdx.x;
    int total = R * NPAD;
    if (idx >= total) return;
    int n = idx % NPAD;
    int r = idx / NPAD;
    float acc = 0.f;
    if (n < N) {
        const float* ap = A + r * KA;
        for (int k = 0; k < Krows; ++k) acc += ap[k] * W[k * ldW + n];
    }
    out[idx] = acc;
}

__global__ void pad_bias(const float* __restrict__ b, float* __restrict__ bp, int N)
{
    int idx = blockIdx.x * blockDim.x + threadIdx.x;
    if (idx >= NPAD) return;
    bp[idx] = (idx < N) ? b[idx] : 0.f;
}

// H1[r][n] = relu(U[b,j] + V[b,i] + Qc[b] + b1), f16, padded rows/cols -> 0
__global__ void compose_h1(const float* __restrict__ U, const float* __restrict__ V,
                           const float* __restrict__ Qc, const float* __restrict__ b1p,
                           _Float16* __restrict__ H)
{
    int idx = blockIdx.x * blockDim.x + threadIdx.x;   // < MPAD*NPAD = 82M < 2^31
    int n = idx % NPAD;
    int r = idx / NPAD;
    if (r >= MPAD) return;
    float v = 0.f;
    if (r < 40000) {
        int b  = r / 625;
        int pr = r % 625;
        int i  = pr / 25;
        int j  = pr % 25;
        v = U[(b * 25 + j) * NPAD + n] + V[(b * 25 + i) * NPAD + n]
          + Qc[b * NPAD + n] + b1p[n];
        v = v > 0.f ? v : 0.f;
    }
    H[idx] = (_Float16)v;
}

// Transpose + convert fp32 W[K][N] -> f16 Wt[NPAD][KPAD], zero-padded
__global__ void prep_wt(const float* __restrict__ W, int K, int N,
                        _Float16* __restrict__ Wt)
{
    int idx = blockIdx.x * blockDim.x + threadIdx.x;   // NPAD*KPAD = 4M
    if (idx >= NPAD * KPAD) return;
    int k = idx % KPAD;
    int n = idx / KPAD;
    _Float16 v = (_Float16)0.f;
    if (k < K && n < N) v = (_Float16)W[k * N + n];
    Wt[idx] = v;
}

// ---------------------------------------------------------------------------
// WMMA GEMM: C[M][NPAD] = relu(A[M][KPAD] x Wt^T + bias), f16 in/out, f32 acc.
// Wave tile 16x128 (8 accumulators), block = 8 waves -> 128x128 tile.
// B operand staged in LDS once per block (async global->LDS when available),
// double-buffered in K-chunks of 64; A streamed per-wave from global/L2.
// Fragment layouts per CDNA5 ISA 7.12.2:
//   A 16x32 f16 : lane L<16 row M=L, K = {0..7,16..23}; lane 16+L same row,
//                 K = {8..15,24..31}
//   B 32x16 f16 : lane c<16 col N=c, K = {0..7,16..23}; lane 16+c K = {8..15,24..31}
//   C 16x16 f32 : VGPR i -> M = i + 8*(lane>=16), N = lane&15
// ---------------------------------------------------------------------------
__global__ __launch_bounds__(256)
void gemm_wmma_relu(const _Float16* __restrict__ A, const _Float16* __restrict__ Wt,
                    const float* __restrict__ bias, _Float16* __restrict__ C)
{
    __shared__ _Float16 Bsh[2][128][BSTRIDE];

    const int tid  = threadIdx.x;
    const int lane = tid & 31;
    const int wave = tid >> 5;
    const int m0 = blockIdx.x * 128 + wave * 16;
    const int n0 = blockIdx.y * 128;
    const int lh = lane & 15;   // row (A) / col (B) within 16-wide tile
    const int lu = lane >> 4;   // half-wave select

    // Stage one 128-col x 64-K chunk of B into LDS buffer `buf`.
    // 1024 x 16B chunks, 4 per thread.
    auto stage = [&](int kc, int buf) {
        #pragma unroll
        for (int i = 0; i < 4; ++i) {
            int c   = tid + i * 256;      // 0..1023
            int row = c >> 3;             // B column 0..127
            int k8  = (c & 7) * 8;        // K offset within chunk
            const _Float16* src = Wt + (long)(n0 + row) * KPAD + kc * 64 + k8;
            copy16_g2l(src, &Bsh[buf][row][k8]);
        }
    };

    v8f acc[8] = {};
    const _Float16* arow = A + (long)(m0 + lh) * KPAD + lu * 8;

    stage(0, 0);
    wait_async_all();
    __syncthreads();

    const int NCHUNK = KPAD / 64;   // 32
    for (int kc = 0; kc < NCHUNK; ++kc) {
        const int cur = kc & 1;
        if (kc + 1 < NCHUNK) stage(kc + 1, cur ^ 1);

        const _Float16* ab = arow + kc * 64;
        #pragma unroll
        for (int s = 0; s < 2; ++s) {           // two K=32 WMMA steps per chunk
            v8h a0 = *(const v8h*)(ab + s * 32);        // K = base+0..7
            v8h a1 = *(const v8h*)(ab + s * 32 + 16);   // K = base+16..23
            v16h a;
            #pragma unroll
            for (int e = 0; e < 8; ++e) { a[e] = a0[e]; a[e + 8] = a1[e]; }

            #pragma unroll
            for (int t = 0; t < 8; ++t) {
                const _Float16* bp = &Bsh[cur][t * 16 + lh][s * 32 + lu * 8];
                v8h b0 = *(const v8h*)bp;
                v8h b1 = *(const v8h*)(bp + 16);
                v16h b;
                #pragma unroll
                for (int e = 0; e < 8; ++e) { b[e] = b0[e]; b[e + 8] = b1[e]; }
                acc[t] = __builtin_amdgcn_wmma_f32_16x16x32_f16(
                    false, a, false, b, (short)0, acc[t], false, false);
            }
        }

        wait_async_all();   // next chunk's B copy complete (this wave's share)
        __syncthreads();    // all waves done reading cur + all copies landed
    }

    #pragma unroll
    for (int t = 0; t < 8; ++t) {
        int col = n0 + t * 16 + lh;
        float bv = bias[col];
        #pragma unroll
        for (int i = 0; i < 8; ++i) {
            int row = m0 + i + lu * 8;
            float v = acc[t][i] + bv;
            v = v > 0.f ? v : 0.f;
            C[(long)row * NPAD + col] = (_Float16)v;
        }
    }
}

// Sum-pool over the 625 pairs per batch element
__global__ void pool_pairs(const _Float16* __restrict__ H, float* __restrict__ pooled)
{
    int idx = blockIdx.x * blockDim.x + threadIdx.x;   // 64*2000
    if (idx >= 64 * 2000) return;
    int n = idx % 2000;
    int b = idx / 2000;
    const _Float16* p = H + (long)b * 625 * NPAD + n;
    float s = 0.f;
    for (int pr = 0; pr < 625; ++pr) s += (float)p[(long)pr * NPAD];
    pooled[idx] = s;
}

// f-MLP dense layer, fp32, 64 rows (tiny)
__global__ void fc(const float* __restrict__ in, const float* __restrict__ W,
                   const float* __restrict__ b, float* __restrict__ out,
                   int R, int K, int N, int relu)
{
    int idx = blockIdx.x * blockDim.x + threadIdx.x;
    if (idx >= R * N) return;
    int n = idx % N;
    int r = idx / N;
    float acc = b[n];
    const float* ip = in + r * K;
    for (int k = 0; k < K; ++k) acc += ip[k] * W[k * N + n];
    if (relu) acc = acc > 0.f ? acc : 0.f;
    out[idx] = acc;
}

__global__ void logsoftmax10(const float* __restrict__ in, float* __restrict__ out)
{
    int r = blockIdx.x * blockDim.x + threadIdx.x;
    if (r >= 64) return;
    const float* p = in + r * 10;
    float mx = p[0];
    for (int i = 1; i < 10; ++i) mx = fmaxf(mx, p[i]);
    float s = 0.f;
    for (int i = 0; i < 10; ++i) s += expf(p[i] - mx);
    float ls = logf(s);
    for (int i = 0; i < 10; ++i) out[r * 10 + i] = p[i] - mx - ls;
}

// ---------------------------------------------------------------------------
extern "C" void kernel_launch(void* const* d_in, const int* in_sizes, int n_in,
                              void* d_out, int out_size, void* d_ws, size_t ws_size,
                              hipStream_t stream)
{
    const float* img = (const float*)d_in[0];
    const float* q   = (const float*)d_in[1];
    const float* conv_w[4]; const float* conv_b[4];
    const float* bn_g[4];   const float* bn_b[4];
    const float* g_w[4];    const float* g_b[4];
    const float* f_w[5];    const float* f_b[5];
    for (int i = 0; i < 4; ++i) {
        conv_w[i] = (const float*)d_in[2 + i];
        conv_b[i] = (const float*)d_in[6 + i];
        bn_g[i]   = (const float*)d_in[10 + i];
        bn_b[i]   = (const float*)d_in[14 + i];
        g_w[i]    = (const float*)d_in[18 + i];
        g_b[i]    = (const float*)d_in[22 + i];
    }
    for (int i = 0; i < 5; ++i) {
        f_w[i] = (const float*)d_in[26 + i];
        f_b[i] = (const float*)d_in[31 + i];
    }

    char* ws = (char*)d_ws;
    size_t off = 0;
    auto carve = [&](size_t bytes) -> void* {
        void* p = ws + off;
        off += (bytes + 255) & ~(size_t)255;
        return p;
    };

    float*    convA  = (float*)   carve(sizeof(float) * 64 * 32 * 40 * 40);
    float*    convB  = (float*)   carve(sizeof(float) * 64 * 64 * 20 * 20);
    float*    mean   = (float*)   carve(sizeof(float) * 256);
    float*    istd   = (float*)   carve(sizeof(float) * 256);
    float*    feat   = (float*)   carve(sizeof(float) * 1600 * 258);
    float*    U      = (float*)   carve(sizeof(float) * 1600 * NPAD);
    float*    V      = (float*)   carve(sizeof(float) * 1600 * NPAD);
    float*    Qc     = (float*)   carve(sizeof(float) * 64 * NPAD);
    float*    bpad   = (float*)   carve(sizeof(float) * NPAD);
    _Float16* Wt     = (_Float16*)carve(sizeof(_Float16) * (size_t)NPAD * KPAD);
    _Float16* Ha     = (_Float16*)carve(sizeof(_Float16) * (size_t)MPAD * NPAD);
    _Float16* Hb     = (_Float16*)carve(sizeof(_Float16) * (size_t)MPAD * NPAD);
    float*    pooled = (float*)   carve(sizeof(float) * 64 * 2000);
    float*    fa     = (float*)   carve(sizeof(float) * 64 * 2000);
    float*    fb     = (float*)   carve(sizeof(float) * 64 * 2000);

    const int T = 256;
    auto blocks = [](long n) { return (int)((n + 255) / 256); };

    // ---- conv stem: (in, out) ping-pong ----
    const int   Cin[4]  = {3, 32, 64, 128};
    const int   Cout[4] = {32, 64, 128, 256};
    const int   Hin[4]  = {80, 40, 20, 10};
    const int   Hout[4] = {40, 20, 10, 5};
    const float* cin[4] = {img, convA, convB, convA};
    float*      cout[4] = {convA, convB, convA, convB};
    for (int s = 0; s < 4; ++s) {
        int HW = Hout[s] * Hout[s];
        long tot = (long)64 * Cout[s] * HW;
        conv3x3_s2<<<blocks(tot), T, 0, stream>>>(cin[s], conv_w[s], conv_b[s], cout[s],
                                                  64, Cin[s], Hin[s], Cout[s], Hout[s]);
        bn_stats<<<Cout[s], T, 0, stream>>>(cout[s], mean, istd, 64, Cout[s], HW);
        bn_relu<<<blocks(tot), T, 0, stream>>>(cout[s], mean, istd, bn_g[s], bn_b[s],
                                               64, Cout[s], HW);
    }

    // ---- features + coords ----
    build_feat<<<blocks(64 * 25 * 258), T, 0, stream>>>(convB, feat);

    // ---- factorized g layer 1 ----
    gemm_slice<<<blocks((long)1600 * NPAD), T, 0, stream>>>(feat, 1600, 258, 258,
                                                            g_w[0], 2000, 2000, U);
    gemm_slice<<<blocks((long)1600 * NPAD), T, 0, stream>>>(feat, 1600, 258, 258,
                                                            g_w[0] + 258 * 2000, 2000, 2000, V);
    gemm_slice<<<blocks((long)64 * NPAD), T, 0, stream>>>(q, 64, 11, 11,
                                                          g_w[0] + 516 * 2000, 2000, 2000, Qc);
    pad_bias<<<blocks(NPAD), T, 0, stream>>>(g_b[0], bpad, 2000);
    compose_h1<<<blocks((long)MPAD * NPAD), T, 0, stream>>>(U, V, Qc, bpad, Ha);

    // ---- g layers 2..4: WMMA GEMMs ----
    _Float16* Hsrc = Ha;
    _Float16* Hdst = Hb;
    dim3 ggrid(MPAD / 128, NPAD / 128, 1);
    for (int l = 1; l < 4; ++l) {
        prep_wt<<<blocks((long)NPAD * KPAD), T, 0, stream>>>(g_w[l], 2000, 2000, Wt);
        pad_bias<<<blocks(NPAD), T, 0, stream>>>(g_b[l], bpad, 2000);
        gemm_wmma_relu<<<ggrid, T, 0, stream>>>(Hsrc, Wt, bpad, Hdst);
        _Float16* tmp = Hsrc; Hsrc = Hdst; Hdst = tmp;
    }
    // final g output now in Hsrc (= Hb after 3 layers)

    // ---- pair sum-pool ----
    pool_pairs<<<blocks((long)64 * 2000), T, 0, stream>>>(Hsrc, pooled);

    // ---- f-MLP ----
    fc<<<blocks((long)64 * 2000), T, 0, stream>>>(pooled, f_w[0], f_b[0], fa, 64, 2000, 2000, 1);
    fc<<<blocks((long)64 * 1000), T, 0, stream>>>(fa, f_w[1], f_b[1], fb, 64, 2000, 1000, 1);
    fc<<<blocks((long)64 * 500),  T, 0, stream>>>(fb, f_w[2], f_b[2], fa, 64, 1000, 500, 1);
    fc<<<blocks((long)64 * 100),  T, 0, stream>>>(fa, f_w[3], f_b[3], fb, 64, 500, 100, 1);
    fc<<<blocks((long)64 * 10),   T, 0, stream>>>(fb, f_w[4], f_b[4], fa, 64, 100, 10, 0);

    logsoftmax10<<<1, 64, 0, stream>>>(fa, (float*)d_out);
}
